// EntropicOTQuantileRegression_2070174237088
// MI455X (gfx1250) — matrix operand order repaired and checked by reading, not compiled, not run on hardware
//
#include <hip/hip_runtime.h>

#define NTOT 1024
#define MTOT 1024
#define DXC  64
#define DYC  16
#define HC   128
#define EPSC 1e-7f
#define MSPLIT 8
#define MT_PER_BLOCK ((MTOT / 16) / MSPLIT)   // 8 m-tiles of 16 per block

typedef __attribute__((ext_vector_type(8)))  float  v8f;
typedef __attribute__((ext_vector_type(16))) __bf16 v16bf;

struct alignas(16) U128 { unsigned int x, y, z, w; };
union FragU { U128 q[2]; v16bf v; };

// ---- LDS layout (bytes) ----
#define OFF_W2T 0                       // 128*128 bf16 = 32768
#define OFF_A   (OFF_W2T + 32768)       // 256*128 bf16 = 65536
#define OFF_HX  (OFF_A   + 65536)       // 16*128 f32   = 8192
#define OFF_HU  (OFF_HX  + 8192)        // 16*128 f32   = 8192
#define OFF_B2  (OFF_HU  + 8192)        // 128 f32      = 512
#define OFF_W3  (OFF_B2  + 512)         // 128 f32      = 512
#define OFF_Y   (OFF_W3  + 512)         // 16*16 f32    = 1024
#define OFF_U   (OFF_Y   + 1024)        // 16*16 f32    = 1024
#define OFF_PHI (OFF_U   + 1024)        // 16*16 f32    = 1024
#define OFF_RED (OFF_PHI + 1024)        // 2*16 f32     = 128
#define SMEM_BYTES (OFF_RED + 128)

#define LOG2E 1.44269504088896f
#define LN2   0.69314718055995f

static __device__ __forceinline__ unsigned short f2bf(float f) {
    union { float f; unsigned u; } v; v.f = f;
    unsigned r = v.u + 0x7FFFu + ((v.u >> 16) & 1u);   // round-to-nearest-even
    return (unsigned short)(r >> 16);
}

// Raw-hardware exp: v_exp_f32 with no denorm fixup (arg <= 0 paths flush to 0, which
// is exactly the desired behavior for logsumexp weights).
static __device__ __forceinline__ float fast_exp(float x) {
    return __builtin_amdgcn_exp2f(x * LOG2E);
}

// Numerically stable softplus on raw v_exp_f32 / v_log_f32:
//   softplus(x) = max(x,0) + ln2 * log2(1 + 2^(-|x|*log2e))
// log argument is in [1,2] -> never denormal, so the raw v_log_f32 is exact enough.
static __device__ __forceinline__ float softplus_f(float x) {
    float e = __builtin_amdgcn_exp2f(-fabsf(x) * LOG2E);
    return fmaxf(x, 0.f) + LN2 * __builtin_amdgcn_logf(1.0f + e);
}

// hx[n][k] = b1[k] + sum_d X[n][d]*W1[d][k];  hu[m][k] = sum_d U[m][d]*W1[DX+d][k]
__global__ void ot_prep(const float* __restrict__ X, const float* __restrict__ U,
                        const float* __restrict__ W1, const float* __restrict__ b1,
                        float* __restrict__ hx, float* __restrict__ hu) {
    int idx = blockIdx.x * blockDim.x + threadIdx.x;   // 0 .. 2*NTOT*HC-1
    int half = idx / (NTOT * HC);
    int r = idx - half * (NTOT * HC);
    int n = r >> 7, k = r & (HC - 1);
    if (half == 0) {
        float acc = b1[k];
        #pragma unroll 8
        for (int d = 0; d < DXC; ++d) acc += X[n * DXC + d] * W1[d * HC + k];
        hx[r] = acc;
    } else {
        float acc = 0.f;
        #pragma unroll
        for (int d = 0; d < DYC; ++d) acc += U[n * DYC + d] * W1[(DXC + d) * HC + k];
        hu[r] = acc;
    }
}

__global__ void ot_main(const float* __restrict__ hx, const float* __restrict__ hu,
                        const float* __restrict__ Y,  const float* __restrict__ U,
                        const float* __restrict__ W2, const float* __restrict__ b2,
                        const float* __restrict__ W3, const float* __restrict__ b3,
                        float2* __restrict__ partials) {
    extern __shared__ char smem[];
    unsigned short* sW2T = (unsigned short*)(smem + OFF_W2T);   // [kout][h] bf16
    unsigned short* sA   = (unsigned short*)(smem + OFF_A);     // [pair][h] bf16
    float* sHx  = (float*)(smem + OFF_HX);
    float* sHu  = (float*)(smem + OFF_HU);
    float* sB2  = (float*)(smem + OFF_B2);
    float* sW3  = (float*)(smem + OFF_W3);
    float* sY   = (float*)(smem + OFF_Y);
    float* sU   = (float*)(smem + OFF_U);
    float* sPhi = (float*)(smem + OFF_PHI);
    float* sRmax = (float*)(smem + OFF_RED);
    float* sRsum = (float*)(smem + OFF_RED + 64);

    const int tid    = threadIdx.x;      // 0..255
    const int ntile  = blockIdx.x;       // 0..63
    const int mchunk = blockIdx.y;       // 0..MSPLIT-1
    const int n0 = ntile * 16;
    const int lane = tid & 31;
    const int wave = tid >> 5;           // 0..7
    const int ln15 = lane & 15;

    // Stage W2 transposed as bf16: sW2T[kout*H + h] = W2[h][kout]
    for (int i = tid; i < HC * HC; i += 256) {
        int h = i >> 7, k = i & (HC - 1);
        sW2T[k * HC + h] = f2bf(W2[i]);
    }
    if (tid < HC) { sB2[tid] = b2[tid]; sW3[tid] = W3[tid]; }
    for (int i = tid; i < 16 * HC; i += 256)
        sHx[i] = hx[(n0 + (i >> 7)) * HC + (i & (HC - 1))];
    if (tid < 16 * DYC) sY[tid] = Y[(n0 + tid / DYC) * DYC + tid % DYC];
    if (tid < 16) { sRmax[tid] = -1e30f; sRsum[tid] = 0.f; }
    __syncthreads();

    const float b3v = b3[0];

    for (int mt = 0; mt < MT_PER_BLOCK; ++mt) {
        const int m0 = (mchunk * MT_PER_BLOCK + mt) * 16;
        for (int i = tid; i < 16 * HC; i += 256)
            sHu[i] = hu[(m0 + (i >> 7)) * HC + (i & (HC - 1))];
        if (tid < 16 * DYC) sU[tid] = U[(m0 + tid / DYC) * DYC + tid % DYC];
        // Prefetch next m-tile's hu panel into cache (global_prefetch_b8).
        if (mt + 1 < MT_PER_BLOCK) {
            const int m1 = (mchunk * MT_PER_BLOCK + mt + 1) * 16;
            __builtin_prefetch(&hu[(m1 + (tid >> 4)) * HC + (tid & 15) * 8], 0, 3);
        }
        __syncthreads();

        // h1 = softplus(hx + hu) -> bf16 A panel.  pair = tid, n_local = tid>>4, m_local = tid&15
        {
            const float* hxr = sHx + (tid >> 4) * HC;
            const float* hur = sHu + (tid & 15) * HC;
            unsigned short* ar = sA + tid * HC;
            #pragma unroll 4
            for (int k = 0; k < HC; ++k)
                ar[k] = f2bf(softplus_f(hxr[k] + hur[k]));
        }
        __syncthreads();

        // GEMM h1@W2 (+b2), softplus, dot with W3 -> phi.
        // Wave owns n_local rows {2w, 2w+1}; A fragments for both row tiles stay resident,
        // B fragments are loaded once per column tile and reused by both row tiles.
        {
            // Load A fragments: fa[rsel][kc], 16-bit A 16x32 layout
            FragU fa[2][4];
            #pragma unroll
            for (int rsel = 0; rsel < 2; ++rsel) {
                const int rt = wave * 2 + rsel;
                #pragma unroll
                for (int kc = 0; kc < 4; ++kc) {
                    const char* pa = (const char*)sA
                        + (rt * 16 + ln15) * (HC * 2) + kc * 64 + ((lane >= 16) ? 16 : 0);
                    fa[rsel][kc].q[0] = *(const U128*)pa;
                    fa[rsel][kc].q[1] = *(const U128*)(pa + 32);
                }
            }

            float psum[2][8];
            #pragma unroll
            for (int rsel = 0; rsel < 2; ++rsel)
                #pragma unroll
                for (int r = 0; r < 8; ++r) psum[rsel][r] = 0.f;

            // Keep this loop ROLLED: full unroll makes the register allocator spill
            // the WMMA accumulators to scratch (observed in round 3).
            #pragma unroll 1
            for (int ct = 0; ct < 8; ++ct) {
                const int col = ct * 16 + ln15;      // output channel of this lane
                // B fragments for all 4 K-chunks of this column tile
                FragU fb[4];
                #pragma unroll
                for (int kc = 0; kc < 4; ++kc) {
                    const char* pb = (const char*)sW2T
                        + col * (HC * 2) + kc * 64 + ((lane >= 16) ? 16 : 0);
                    fb[kc].q[0] = *(const U128*)pb;
                    fb[kc].q[1] = *(const U128*)(pb + 32);
                }
                const float b2c = sB2[col];
                const float w3c = sW3[col];
                #pragma unroll
                for (int rsel = 0; rsel < 2; ++rsel) {
                    v8f acc = { b2c, b2c, b2c, b2c, b2c, b2c, b2c, b2c };
                    #pragma unroll
                    for (int kc = 0; kc < 4; ++kc)
                        acc = __builtin_amdgcn_wmma_f32_16x16x32_bf16(
                            false, fa[rsel][kc].v, false, fb[kc].v, (short)0, acc,
                            false, false);
                    #pragma unroll
                    for (int r = 0; r < 8; ++r)
                        psum[rsel][r] += softplus_f(acc[r]) * w3c;
                }
            }

            // reduce across the 16 lanes holding the 16 columns (wave32: width-16 xor shuffles)
            #pragma unroll
            for (int rsel = 0; rsel < 2; ++rsel) {
                #pragma unroll
                for (int r = 0; r < 8; ++r) {
                    float v = psum[rsel][r];
                    for (int mask = 1; mask < 16; mask <<= 1) v += __shfl_xor(v, mask, 16);
                    psum[rsel][r] = v;
                }
                if (ln15 == 0) {
                    const int rt = wave * 2 + rsel;
                    const int half = (lane >> 4) * 8;  // lanes 0-15 -> rows 0-7, 16-31 -> 8-15
                    #pragma unroll
                    for (int r = 0; r < 8; ++r)
                        sPhi[rt * 16 + half + r] = psum[rsel][r] + b3v;  // sPhi[n_local][m_local]
                }
            }
        }
        __syncthreads();

        // online logsumexp over this m-tile: thread -> (n_local = tid>>4, m_local = tid&15)
        {
            const int nl = tid >> 4, ml = tid & 15;
            float c = 0.f;
            #pragma unroll
            for (int d = 0; d < DYC; ++d) c += sY[nl * DYC + d] * sU[ml * DYC + d];
            float s = (c - sPhi[nl * 16 + ml]) * (1.0f / EPSC);
            float tmax = s;
            for (int mask = 1; mask < 16; mask <<= 1)
                tmax = fmaxf(tmax, __shfl_xor(tmax, mask, 16));
            float e = fast_exp(s - tmax);
            for (int mask = 1; mask < 16; mask <<= 1) e += __shfl_xor(e, mask, 16);
            if (ml == 0) {
                float rm = sRmax[nl], rs = sRsum[nl];
                float nm = fmaxf(rm, tmax);
                sRsum[nl] = rs * fast_exp(rm - nm) + e * fast_exp(tmax - nm);
                sRmax[nl] = nm;
            }
        }
        __syncthreads();
    }

    if (tid < 16)
        partials[(ntile * MSPLIT + mchunk) * 16 + tid] = make_float2(sRmax[tid], sRsum[tid]);
}

__global__ void ot_final(const float2* __restrict__ partials, float* __restrict__ out) {
    int n = blockIdx.x * blockDim.x + threadIdx.x;
    if (n >= NTOT) return;
    int ntile = n >> 4, nl = n & 15;
    float gm = -1e30f, gs = 0.f;
    #pragma unroll
    for (int mc = 0; mc < MSPLIT; ++mc) {
        float2 p = partials[(ntile * MSPLIT + mc) * 16 + nl];
        float nm = fmaxf(gm, p.x);
        gs = gs * fast_exp(gm - nm) + p.y * fast_exp(p.x - nm);
        gm = nm;
    }
    // gs >= 1 here (the max term contributes exp(0)=1), so raw v_log_f32 is safe.
    out[n] = EPSC * (gm + LN2 * __builtin_amdgcn_logf(gs) - __logf((float)MTOT));
}

extern "C" void kernel_launch(void* const* d_in, const int* in_sizes, int n_in,
                              void* d_out, int out_size, void* d_ws, size_t ws_size,
                              hipStream_t stream) {
    (void)in_sizes; (void)n_in; (void)out_size; (void)ws_size;
    const float* X  = (const float*)d_in[0];
    const float* U  = (const float*)d_in[1];
    const float* Y  = (const float*)d_in[2];
    const float* W1 = (const float*)d_in[3];
    const float* b1 = (const float*)d_in[4];
    const float* W2 = (const float*)d_in[5];
    const float* b2 = (const float*)d_in[6];
    const float* W3 = (const float*)d_in[7];
    const float* b3 = (const float*)d_in[8];

    float*  hx = (float*)d_ws;                         // NTOT*HC
    float*  hu = hx + NTOT * HC;                       // MTOT*HC
    float2* partials = (float2*)(hu + MTOT * HC);      // 64*MSPLIT*16 float2

    ot_prep<<<(2 * NTOT * HC) / 256, 256, 0, stream>>>(X, U, W1, b1, hx, hu);

    dim3 grid(NTOT / 16, MSPLIT);
    ot_main<<<grid, 256, SMEM_BYTES, stream>>>(hx, hu, Y, U, W2, b2, W3, b3, partials);

    ot_final<<<NTOT / 256, 256, 0, stream>>>(partials, (float*)d_out);
}
